// Decoder1_14370960572996
// MI455X (gfx1250) — matrix operand order repaired and checked by reading, not compiled, not run on hardware
//
#include <hip/hip_runtime.h>
#include <math.h>

typedef __attribute__((ext_vector_type(16))) __bf16 v16bf;
typedef __attribute__((ext_vector_type(8)))  float  v8f;
typedef __attribute__((ext_vector_type(2)))  float  v2f;

union BFrag { v16bf v; uint4 q[2]; };

#ifndef __has_builtin
#define __has_builtin(x) 0
#endif

static __device__ __forceinline__ unsigned short f2bf(float f) {
    unsigned int b = __float_as_uint(f);
    unsigned int r = b + 0x7FFFu + ((b >> 16) & 1u);   // round-to-nearest-even
    return (unsigned short)(r >> 16);
}

static __device__ __forceinline__ unsigned int f2key(float f) {
    unsigned int x = __float_as_uint(f);
    return ((int)x < 0) ? ~x : (x | 0x80000000u);       // monotonic float->u32
}

// Generic LDS address -> 32-bit DS offset (low 32 bits of the LDS aperture).
static __device__ __forceinline__ unsigned int lds_off32(const void* p) {
    return (unsigned int)(size_t)p;
}

// GLOBAL_LOAD_ASYNC_TO_LDS_B128, GVS mode: mem = SADDR + VADDR32, LDS = VDST vgpr.
// Tracked by ASYNCcnt; no VGPR data round-trip.
static __device__ __forceinline__ void async_load_b128(
    unsigned int lds_addr, unsigned int goff, const void* sbase) {
    asm volatile("global_load_async_to_lds_b128 %0, %1, %2"
                 :
                 : "v"(lds_addr), "v"(goff), "s"(sbase)
                 : "memory");
}

static __device__ __forceinline__ void wait_async0() {
#if __has_builtin(__builtin_amdgcn_s_wait_asynccnt)
    __builtin_amdgcn_s_wait_asynccnt(0);
#else
    asm volatile("s_wait_asynccnt 0x0" ::: "memory");
#endif
}

// ---------------------------------------------------------------------------
// Kernel 0: per-(b,p) L2-normalize over C=384, transpose to [B][HW][C] bf16.
// grid (32, 64, 2), block (32, 8).  z selects cur/goal.
// ---------------------------------------------------------------------------
__global__ __launch_bounds__(256) void normalize_kernel(
    const float* __restrict__ cur, const float* __restrict__ goal,
    unsigned short* __restrict__ curN, unsigned short* __restrict__ goalN)
{
    __shared__ float red[8][32];
    __shared__ float scale_s[32];
    __shared__ float tile[32][33];

    const int tx = threadIdx.x, ty = threadIdx.y;
    const int b  = blockIdx.y;
    const int p0 = blockIdx.x * 32;
    const int p  = p0 + tx;

    const float* src = (blockIdx.z == 0 ? cur : goal) + (size_t)b * 384 * 1024;
    unsigned short* dst = (blockIdx.z == 0 ? curN : goalN) + (size_t)b * 1024 * 384;

    float ss = 0.f;
    for (int c = ty; c < 384; c += 8) {
        float v = src[(size_t)c * 1024 + p];
        ss += v * v;
    }
    red[ty][tx] = ss;
    __syncthreads();
    if (ty == 0) {
        float s = 0.f;
        #pragma unroll
        for (int i = 0; i < 8; ++i) s += red[i][tx];
        scale_s[tx] = 1.0f / fmaxf(sqrtf(s), 1e-12f);
    }
    __syncthreads();

    for (int cb = 0; cb < 384; cb += 32) {
        #pragma unroll
        for (int i = 0; i < 4; ++i) {
            int cl = ty + i * 8;
            tile[cl][tx] = src[(size_t)(cb + cl) * 1024 + p] * scale_s[tx];
        }
        __syncthreads();
        #pragma unroll
        for (int i = 0; i < 4; ++i) {
            int pl = ty + i * 8;
            dst[(size_t)(p0 + pl) * 384 + cb + tx] = f2bf(tile[tx][pl]);
        }
        __syncthreads();
    }
}

// ---------------------------------------------------------------------------
// Kernel 1: score[b] = curN[b] (1024x384) * goalN[b]^T  -> fp32, bf16 WMMA.
// grid (8, 8, 64) tiles of 128x128, block 256 (8 waves, each 32x64).
// LDS: 2 x 128 rows x (96+8) ushort, row stride 208 B (16B aligned, bank-clean)
// Tiles are staged with GLOBAL_LOAD_ASYNC_TO_LDS_B128 (ASYNCcnt).
// ---------------------------------------------------------------------------
__global__ __launch_bounds__(256) void score_gemm_kernel(
    const unsigned short* __restrict__ curN,
    const unsigned short* __restrict__ goalN,
    float* __restrict__ score)
{
    __shared__ unsigned short As[128][104];
    __shared__ unsigned short Bs[128][104];

    const int b    = blockIdx.z;
    const int m0   = blockIdx.y * 128;
    const int n0   = blockIdx.x * 128;
    const int tid  = threadIdx.x;
    const int lane = tid & 31;
    const int wave = tid >> 5;
    const int wm   = wave & 3;       // 4 wave-rows of 32
    const int wn   = wave >> 2;      // 2 wave-cols of 64
    const int g    = lane >> 4;      // lane group
    const int l    = lane & 15;

    const unsigned short* aBase = curN  + ((size_t)b * 1024 + m0) * 384;
    const unsigned short* bBase = goalN + ((size_t)b * 1024 + n0) * 384;

    const unsigned int ldsA0 = lds_off32(&As[0][0]);
    const unsigned int ldsB0 = lds_off32(&Bs[0][0]);

    v8f acc[2][4];
    const v8f vz = {0.f, 0.f, 0.f, 0.f, 0.f, 0.f, 0.f, 0.f};
    #pragma unroll
    for (int mi = 0; mi < 2; ++mi)
        #pragma unroll
        for (int ni = 0; ni < 4; ++ni) acc[mi][ni] = vz;

    for (int kc = 0; kc < 4; ++kc) {
        const int kb = kc * 96;
        // async stage A/B tiles: 128 rows x 96 ch (192 B = 12 x B128 per row)
        #pragma unroll
        for (int q = tid, it = 0; it < 6; ++it, q += 256) {
            int r = q / 12, cq = q % 12;
            unsigned int loff = (unsigned int)(r * 208 + cq * 16);
            unsigned int goff = (unsigned int)((r * 384 + kb + cq * 8) * 2);
            async_load_b128(ldsA0 + loff, goff, aBase);
            async_load_b128(ldsB0 + loff, goff, bBase);
        }
        wait_async0();
        __syncthreads();

        #pragma unroll
        for (int ks = 0; ks < 3; ++ks) {
            BFrag af[2], bfr[4];
            #pragma unroll
            for (int mi = 0; mi < 2; ++mi) {
                const unsigned short* ap = &As[wm * 32 + mi * 16 + l][ks * 32];
                af[mi].q[0] = *(const uint4*)(ap + g * 8);        // K = g*8..g*8+7
                af[mi].q[1] = *(const uint4*)(ap + 16 + g * 8);   // K = 16+g*8..
            }
            #pragma unroll
            for (int ni = 0; ni < 4; ++ni) {
                const unsigned short* bp = &Bs[wn * 64 + ni * 16 + l][ks * 32];
                bfr[ni].q[0] = *(const uint4*)(bp + g * 8);       // K = g*8..g*8+7
                bfr[ni].q[1] = *(const uint4*)(bp + 16 + g * 8);  // K = 16+g*8..
            }
            #pragma unroll
            for (int mi = 0; mi < 2; ++mi)
                #pragma unroll
                for (int ni = 0; ni < 4; ++ni)
                    acc[mi][ni] = __builtin_amdgcn_wmma_f32_16x16x32_bf16(
                        false, af[mi].v, false, bfr[ni].v,
                        (short)0, acc[mi][ni], false, false);
        }
        __syncthreads();
    }

    float* outp = score + ((size_t)b << 20);
    #pragma unroll
    for (int mi = 0; mi < 2; ++mi)
        #pragma unroll
        for (int ni = 0; ni < 4; ++ni) {
            const int col = n0 + wn * 64 + ni * 16 + l;
            #pragma unroll
            for (int vv = 0; vv < 8; ++vv) {
                const int row = m0 + wm * 32 + mi * 16 + g * 8 + vv;
                outp[((size_t)row << 10) + col] = acc[mi][ni][vv];
            }
        }
}

// ---------------------------------------------------------------------------
// Kernel 2: per-batch top-1024 of 1M scores, tie-break by smallest index,
// final set sorted ascending by flat index, packed as (row, col, value).
// grid 64, block 1024.
// ---------------------------------------------------------------------------
__global__ __launch_bounds__(1024) void topk_feats_kernel(
    const float* __restrict__ score, float* __restrict__ feats)
{
    __shared__ unsigned int hist[4096];
    __shared__ unsigned int sel[1024];
    __shared__ unsigned int tie[4096];
    __shared__ unsigned int sc[8];   // 0:b0 1:b1 2:b2 3:G(cum) 4:T 5:selCnt 6:tieCnt

    const int b   = blockIdx.x;
    const int tid = threadIdx.x;
    const float* sb = score + ((size_t)b << 20);
    const unsigned int KTOP = 1024u;
    const int NTOT = 1 << 20;

    // ---- level 0: top 12 bits ----
    for (int i = tid; i < 4096; i += 1024) hist[i] = 0;
    __syncthreads();
    for (int i = tid; i < NTOT; i += 1024)
        atomicAdd(&hist[f2key(sb[i]) >> 20], 1u);
    __syncthreads();
    if (tid == 0) {
        unsigned int cum = 0;
        for (int bb = 4095; bb >= 0; --bb) {
            unsigned int h = hist[bb];
            if (cum + h >= KTOP) { sc[0] = (unsigned)bb; sc[3] = cum; break; }
            cum += h;
        }
    }
    __syncthreads();
    const unsigned int b0 = sc[0];

    // ---- level 1: next 12 bits within bin b0 ----
    for (int i = tid; i < 4096; i += 1024) hist[i] = 0;
    __syncthreads();
    for (int i = tid; i < NTOT; i += 1024) {
        unsigned int u = f2key(sb[i]);
        if ((u >> 20) == b0) atomicAdd(&hist[(u >> 8) & 0xFFFu], 1u);
    }
    __syncthreads();
    if (tid == 0) {
        unsigned int cum = sc[3];
        for (int bb = 4095; bb >= 0; --bb) {
            unsigned int h = hist[bb];
            if (cum + h >= KTOP) { sc[1] = (unsigned)bb; sc[3] = cum; break; }
            cum += h;
        }
    }
    __syncthreads();
    const unsigned int top24 = (b0 << 12) | sc[1];

    // ---- level 2: low 8 bits within top-24 prefix ----
    if (tid < 256) hist[tid] = 0;
    __syncthreads();
    for (int i = tid; i < NTOT; i += 1024) {
        unsigned int u = f2key(sb[i]);
        if ((u >> 8) == top24) atomicAdd(&hist[u & 0xFFu], 1u);
    }
    __syncthreads();
    if (tid == 0) {
        unsigned int cum = sc[3];
        for (int bb = 255; bb >= 0; --bb) {
            unsigned int h = hist[bb];
            if (cum + h >= KTOP) { sc[2] = (unsigned)bb; sc[3] = cum; break; }
            cum += h;
        }
        sc[4] = (top24 << 8) | sc[2];
        sc[5] = 0; sc[6] = 0;
    }
    __syncthreads();
    const unsigned int T = sc[4];
    const unsigned int G = sc[3];
    const unsigned int R = KTOP - G;

    // ---- compaction: strictly-above into sel[], ties into tie[] ----
    for (int i = tid; i < NTOT; i += 1024) {
        unsigned int u = f2key(sb[i]);
        if (u > T) {
            unsigned int p = atomicAdd(&sc[5], 1u);
            if (p < 1024u) sel[p] = (unsigned int)i;
        } else if (u == T) {
            unsigned int p = atomicAdd(&sc[6], 1u);
            if (p < 4096u) tie[p] = (unsigned int)i;
        }
    }
    __syncthreads();
    unsigned int tc = sc[6] < 4096u ? sc[6] : 4096u;
    for (int i = tid; i < 4096; i += 1024)
        if ((unsigned int)i >= tc) tie[i] = 0xFFFFFFFFu;
    __syncthreads();

    // ---- bitonic sort tie[] ascending (n=4096) ----
    for (unsigned int k = 2; k <= 4096; k <<= 1)
        for (unsigned int j = k >> 1; j > 0; j >>= 1) {
            for (int e = tid; e < 4096; e += 1024) {
                int p = e ^ (int)j;
                if (p > e) {
                    unsigned int a = tie[e], c = tie[p];
                    bool asc = ((e & k) == 0);
                    if ((a > c) == asc) { tie[e] = c; tie[p] = a; }
                }
            }
            __syncthreads();
        }
    if (tid < (int)R) sel[G + tid] = tie[tid];   // R smallest tie indices
    __syncthreads();

    // ---- bitonic sort sel[] ascending (n=1024) ----
    for (unsigned int k = 2; k <= 1024; k <<= 1)
        for (unsigned int j = k >> 1; j > 0; j >>= 1) {
            int p = tid ^ (int)j;
            if (p > tid) {
                unsigned int a = sel[tid], c = sel[p];
                bool asc = ((tid & k) == 0);
                if ((a > c) == asc) { sel[tid] = c; sel[p] = a; }
            }
            __syncthreads();
        }

    const unsigned int idx = sel[tid];
    float* f = feats + (size_t)b * 3072 + (size_t)tid * 3;
    f[0] = (float)(idx >> 10);
    f[1] = (float)(idx & 1023u);
    f[2] = sb[idx];
}

// ---------------------------------------------------------------------------
// Kernel 3: Y = silu(X*W + bias), M = 64, full-precision fp32 WMMA 16x16x4.
// Each wave computes one 16x16 output tile. block 256 (8 waves).
// ---------------------------------------------------------------------------
__global__ __launch_bounds__(256) void linear_silu_kernel(
    const float* __restrict__ X, const float* __restrict__ W,
    const float* __restrict__ bias, float* __restrict__ Y,
    int Kdim, int N, int do_silu)
{
    const int lane = threadIdx.x & 31;
    const int wave = threadIdx.x >> 5;
    const int wg   = blockIdx.x * 8 + wave;
    const int tiles = 4 * (N >> 4);           // M=64 -> 4 row tiles
    if (wg >= tiles) return;

    const int mi = wg & 3, ni = wg >> 2;
    const int m0 = mi * 16, n0 = ni * 16;
    const int g  = lane >> 4, l = lane & 15;

    v8f acc = {0.f, 0.f, 0.f, 0.f, 0.f, 0.f, 0.f, 0.f};
    const float* xrow = X + (size_t)(m0 + l) * Kdim;
    for (int k = 0; k < Kdim; k += 4) {
        v2f a, bv;
        a.x  = xrow[k + 2 * g];
        a.y  = xrow[k + 2 * g + 1];
        bv.x = W[(size_t)(k + 2 * g) * N + n0 + l];
        bv.y = W[(size_t)(k + 2 * g + 1) * N + n0 + l];
        acc = __builtin_amdgcn_wmma_f32_16x16x4_f32(
            false, a, false, bv, (short)0, acc, false, false);
    }
    const float bb = bias[n0 + l];
    #pragma unroll
    for (int vv = 0; vv < 8; ++vv) {
        float x = acc[vv] + bb;
        if (do_silu) x = x / (1.0f + expf(-x));
        Y[(size_t)(m0 + g * 8 + vv) * N + n0 + l] = x;
    }
}

// ---------------------------------------------------------------------------
// Kernel 4: three 16->3 heads -> action [B,3,3]. 576 threads.
// ---------------------------------------------------------------------------
__global__ void heads_kernel(
    const float* __restrict__ h4,
    const float* __restrict__ Wx, const float* __restrict__ bx,
    const float* __restrict__ Wy, const float* __restrict__ by,
    const float* __restrict__ Wr, const float* __restrict__ br,
    float* __restrict__ out)
{
    int t = threadIdx.x;
    if (t >= 576) return;
    int b = t / 9, rem = t % 9, head = rem / 3, j = rem % 3;
    const float* W = (head == 0) ? Wx : (head == 1) ? Wy : Wr;
    const float* bs = (head == 0) ? bx : (head == 1) ? by : br;
    float s = bs[j];
    const float* h = h4 + (size_t)b * 16;
    #pragma unroll
    for (int i = 0; i < 16; ++i) s += h[i] * W[i * 3 + j];
    out[t] = s;
}

// ---------------------------------------------------------------------------
extern "C" void kernel_launch(void* const* d_in, const int* in_sizes, int n_in,
                              void* d_out, int out_size, void* d_ws, size_t ws_size,
                              hipStream_t stream) {
    (void)in_sizes; (void)n_in; (void)out_size; (void)ws_size;

    const float* cur  = (const float*)d_in[0];
    const float* goal = (const float*)d_in[1];
    const float* W1 = (const float*)d_in[2];  const float* b1 = (const float*)d_in[3];
    const float* W2 = (const float*)d_in[4];  const float* b2 = (const float*)d_in[5];
    const float* W3 = (const float*)d_in[6];  const float* b3 = (const float*)d_in[7];
    const float* W4 = (const float*)d_in[8];  const float* b4 = (const float*)d_in[9];
    const float* Wx = (const float*)d_in[10]; const float* bx = (const float*)d_in[11];
    const float* Wy = (const float*)d_in[12]; const float* by = (const float*)d_in[13];
    const float* Wr = (const float*)d_in[14]; const float* br = (const float*)d_in[15];

    // workspace layout
    unsigned short* curN  = (unsigned short*)d_ws;
    unsigned short* goalN = curN + (size_t)64 * 1024 * 384;
    float* feats = (float*)(goalN + (size_t)64 * 1024 * 384);
    float* h1 = feats + 64 * 3072;
    float* h2 = h1 + 64 * 1024;
    float* h3 = h2 + 64 * 256;
    float* h4 = h3 + 64 * 64;

    float* out   = (float*)d_out;        // action [64,3,3] = 576 floats
    float* score = out + 576;            // score_matrix [64,1024,1024]

    normalize_kernel<<<dim3(32, 64, 2), dim3(32, 8), 0, stream>>>(cur, goal, curN, goalN);
    score_gemm_kernel<<<dim3(8, 8, 64), 256, 0, stream>>>(curN, goalN, score);
    topk_feats_kernel<<<64, 1024, 0, stream>>>(score, feats);
    linear_silu_kernel<<<32, 256, 0, stream>>>(feats, W1, b1, h1, 3072, 1024, 1);
    linear_silu_kernel<<<8, 256, 0, stream>>>(h1, W2, b2, h2, 1024, 256, 1);
    linear_silu_kernel<<<2, 256, 0, stream>>>(h2, W3, b3, h3, 256, 64, 1);
    linear_silu_kernel<<<1, 256, 0, stream>>>(h3, W4, b4, h4, 64, 16, 1);
    heads_kernel<<<1, 576, 0, stream>>>(h4, Wx, bx, Wy, by, Wr, br, out);
}
